// ECPEC_37512244363407
// MI455X (gfx1250) — matrix operand order repaired and checked by reading, not compiled, not run on hardware
//
#include <hip/hip_runtime.h>
#include <math.h>

// ---------------- types ----------------
typedef __bf16 bf16_t;
typedef __attribute__((ext_vector_type(16))) __bf16 v16bf;
typedef __attribute__((ext_vector_type(8)))  __bf16 v8bf;
typedef __attribute__((ext_vector_type(8)))  float  v8f;

#define D_ 512
#define SE_ 256
#define SC_ 512
#define CS_ 8
#define K_ 64
#define NC_ 2

__device__ __forceinline__ v16bf cat8(v8bf lo, v8bf hi) {
  return __builtin_shufflevector(lo, hi, 0,1,2,3,4,5,6,7,8,9,10,11,12,13,14,15);
}
__device__ __forceinline__ float sigm(float x) { return 1.0f / (1.0f + __expf(-x)); }

// low 32 bits of a generic pointer to __shared__ = LDS byte offset on amdgcn
__device__ __forceinline__ unsigned lds_off(const void* p) {
  return (unsigned)(uintptr_t)p;
}
// async global->LDS copy of 32 bytes (two b128 ops; INST_OFFSET applies to both
// the LDS and global addresses, so one address pair covers both halves)
__device__ __forceinline__ void async_copy32(unsigned ldsAddr, const void* gAddr) {
  asm volatile("global_load_async_to_lds_b128 %0, %1, off\n\t"
               "global_load_async_to_lds_b128 %0, %1, off offset:16"
               :: "v"(ldsAddr), "v"(gAddr) : "memory");
}
__device__ __forceinline__ void wait_async4() {
  asm volatile("s_wait_asynccnt 0x4" ::: "memory");
}
__device__ __forceinline__ void wait_async0() {
  asm volatile("s_wait_asynccnt 0x0" ::: "memory");
}

// ---------------- f32 -> bf16 slice convert ----------------
__global__ void __launch_bounds__(256)
cvt_bf16(const float* __restrict__ src, int srcRowLen, int colOff,
         bf16_t* __restrict__ dst, int rows, int cols) {
  int idx = blockIdx.x * 256 + threadIdx.x;
  if (idx >= rows * cols) return;
  int r = idx / cols, c = idx - r * cols;
  dst[idx] = (bf16_t)src[(size_t)r * srcRowLen + colOff + c];
}

__global__ void __launch_bounds__(256)
zero_h(float* h, bf16_t* hb) {
  int i = blockIdx.x * 256 + threadIdx.x;
  if (i < K_ * D_) { h[i] = 0.0f; hb[i] = (bf16_t)0.0f; }
}

// ---------------- generic WMMA GEMM: C = act(A @ W^T + bias) ----------------
// A: bf16, logical M rows, row r at A + r*aRowStride, K contiguous elems.
// W: bf16, N x K row-major. M % 64 == 0, N % 64 == 0, K % 32 == 0.
// act: 0 none, 1 relu, 2 sigmoid. Optional bf16 mirror output Cb.
// Double-buffered: tiles staged global->LDS with async-to-LDS ops (ASYNCcnt),
// prefetch of tile t+1 overlaps WMMA compute of tile t.
__global__ void __launch_bounds__(128)
wmma_gemm(const bf16_t* __restrict__ A, int aRowStride,
          const bf16_t* __restrict__ W,
          const float* __restrict__ bias,
          float* __restrict__ C, bf16_t* __restrict__ Cb,
          int M, int N, int K, int act) {
  __shared__ __align__(16) bf16_t As[2][64][32];
  __shared__ __align__(16) bf16_t Ws[2][64][32];
  int mBase = blockIdx.y * 64;
  int nBase = blockIdx.x * 64;
  int tid = threadIdx.x;
  int wave = tid >> 5, lane = tid & 31;
  int laneHi = lane >> 4;
  v8f acc[4] = {};

  // per-thread staging slot: 32B of one 64x32 tile row-half
  int r = tid >> 1;
  int c = (tid & 1) * 16;
  const bf16_t* aSrc = A + (size_t)(mBase + r) * aRowStride + c;
  const bf16_t* wSrc = W + (size_t)(nBase + r) * K + c;
  unsigned aDst = lds_off(&As[0][r][c]);
  unsigned wDst = lds_off(&Ws[0][r][c]);
  const unsigned bufBytes = 64 * 32 * 2;  // 4096

  int T = K >> 5;
  // prefetch tile 0 into buffer 0 (4 async ops per thread per tile)
  async_copy32(aDst, aSrc);
  async_copy32(wDst, wSrc);

  for (int t = 0; t < T; ++t) {
    int buf = t & 1;
    if (t + 1 < T) {
      unsigned nb = (unsigned)(buf ^ 1) * bufBytes;
      async_copy32(aDst + nb, aSrc + (t + 1) * 32);
      async_copy32(wDst + nb, wSrc + (t + 1) * 32);
      wait_async4();  // newest 4 = prefetch; <=4 means tile t has landed
    } else {
      wait_async0();
    }
    __syncthreads();
    // A fragment: lane holds M=lane&15, K in {base..base+7} U {base+16..base+23}
    int mRow = wave * 16 + (lane & 15);
    int kb = laneHi * 8;
    v16bf af = cat8(*(const v8bf*)&As[buf][mRow][kb],
                    *(const v8bf*)&As[buf][mRow][kb + 16]);
    for (int ns = 0; ns < 4; ++ns) {
      int nRow = ns * 16 + (lane & 15);
      int kg = laneHi * 8;
      v16bf bfg = cat8(*(const v8bf*)&Ws[buf][nRow][kg],
                       *(const v8bf*)&Ws[buf][nRow][kg + 16]);
      acc[ns] = __builtin_amdgcn_wmma_f32_16x16x32_bf16(
          false, af, false, bfg, (short)0, acc[ns], false, false);
    }
    __syncthreads();  // all waves done reading buf before it is prefetched again
  }
  for (int ns = 0; ns < 4; ++ns) {
    int n = nBase + ns * 16 + (lane & 15);
    float bv = bias ? bias[n] : 0.0f;
    for (int i = 0; i < 8; ++i) {
      int m = mBase + wave * 16 + i + laneHi * 8;
      float v = acc[ns][i] + bv;
      if (act == 1) v = fmaxf(v, 0.0f);
      else if (act == 2) v = sigm(v);
      C[(size_t)m * N + n] = v;
      if (Cb) Cb[(size_t)m * N + n] = (bf16_t)v;
    }
  }
}

// ---------------- GRU gate fusion ----------------
__global__ void __launch_bounds__(256)
gru_gate(const float* __restrict__ gx, const float* __restrict__ gh,
         float* __restrict__ h, bf16_t* __restrict__ hb) {
  int idx = blockIdx.x * 256 + threadIdx.x;
  if (idx >= K_ * D_) return;
  int row = idx >> 9, d = idx & (D_ - 1);
  const float* gxr = gx + row * 3 * D_;
  const float* ghr = gh + row * 3 * D_;
  float r = sigm(gxr[d] + ghr[d]);
  float z = sigm(gxr[D_ + d] + ghr[D_ + d]);
  float n = tanhf(gxr[2 * D_ + d] + r * ghr[2 * D_ + d]);
  float hv = (1.0f - z) * n + z * h[idx];
  h[idx] = hv;
  hb[idx] = (bf16_t)hv;
}

// ---------------- phase2: hidden = sigmoid(emoP[j]+chP[k]); 2-class log-softmax ----
__global__ void __launch_bounds__(256)
phase2_kernel(const float* __restrict__ emoP, const float* __restrict__ chP,
              const float* __restrict__ Wo, const float* __restrict__ Wob,
              float* __restrict__ out) {
  int wid = (blockIdx.x * 256 + threadIdx.x) >> 5;
  int lane = threadIdx.x & 31;
  if (wid >= SE_ * K_) return;
  int j = wid >> 6, k = wid & (K_ - 1);
  float s0 = 0.0f, s1 = 0.0f;
  for (int d = lane; d < D_; d += 32) {
    float hv = sigm(emoP[j * D_ + d] + chP[k * D_ + d]);
    s0 += hv * Wo[d];
    s1 += hv * Wo[D_ + d];
  }
  for (int off = 16; off; off >>= 1) {
    s0 += __shfl_xor(s0, off, 32);
    s1 += __shfl_xor(s1, off, 32);
  }
  if (lane == 0) {
    float l0 = s0 + Wob[0], l1 = s1 + Wob[1];
    float mx = fmaxf(l0, l1);
    float lse = mx + logf(expf(l0 - mx) + expf(l1 - mx));
    out[wid * 2 + 0] = l0 - lse;
    out[wid * 2 + 1] = l1 - lse;
  }
}

// ---------------- pair list: row-major nonzero(label_ck) via block scan ----------
__global__ void __launch_bounds__(1024)
pair_scan(const int* __restrict__ lck, int* __restrict__ pj, int* __restrict__ pk,
          int* __restrict__ pcount) {
  __shared__ int sums[1024];
  int tid = threadIdx.x;
  int base = tid * 16;
  int cnt = 0;
  for (int i = 0; i < 16; ++i) cnt += (lck[base + i] != 0);
  sums[tid] = cnt;
  __syncthreads();
  for (int off = 1; off < 1024; off <<= 1) {
    int v = (tid >= off) ? sums[tid - off] : 0;
    __syncthreads();
    sums[tid] += v;
    __syncthreads();
  }
  int pos = sums[tid] - cnt;  // exclusive prefix
  for (int i = 0; i < 16; ++i) {
    int idx = base + i;
    if (lck[idx]) { pj[pos] = idx >> 6; pk[pos] = idx & (K_ - 1); pos++; }
  }
  if (tid == 1023) *pcount = sums[1023];
}

// ---------------- label3 output ----------------
__global__ void __launch_bounds__(256)
label3_kernel(const int* __restrict__ pj, const int* __restrict__ pk,
              const int* __restrict__ l3, float* __restrict__ out3, int M8) {
  int r = blockIdx.x * 256 + threadIdx.x;
  if (r >= M8) return;
  int p = r >> 3, t = r & 7;
  int j = pj[p], k = pk[p];
  int c = k * CS_ + t;
  int lbl = 0;
  for (int m = 0; m < 4; ++m) lbl |= (l3[j * 4 + m] == c);
  out3[r] = (float)lbl;
}

// ---------------- pair kernel: h = relu(emoA[j]+cauB[kt]+nrm*w3c+b3+(e*tok)@D^T)
//                  then 2-class classifier + log-softmax, fused. ----------------
__global__ void __launch_bounds__(128)
pair_kernel(const int* __restrict__ pj, const int* __restrict__ pk,
            const float* __restrict__ emo, const float* __restrict__ cau,
            const float* __restrict__ emoA, const float* __restrict__ cauB,
            const bf16_t* __restrict__ Dw,          // 512x512 bf16 (W3 cols 1025..1536)
            const float* __restrict__ W3w,          // raw 512x1537 (for norm column 1024)
            const float* __restrict__ W3b,
            const float* __restrict__ clsw, const float* __restrict__ clsb,
            float* __restrict__ out2, int M8) {
  __shared__ __align__(16) bf16_t As[16][D_];
  __shared__ float Hs[16][D_];
  __shared__ float segs[16][8];
  __shared__ float nrm[16];
  __shared__ int rJ[16], rC[16];
  int tid = threadIdx.x;
  int r0 = blockIdx.x * 16;
  if (tid < 16) {
    int r = r0 + tid;
    int v = (r < M8);
    int p = v ? (r >> 3) : 0;
    int t = v ? (r & 7) : 0;
    int j = v ? pj[p] : 0;
    int k = v ? pk[p] : 0;
    rJ[tid] = j;
    rC[tid] = k * CS_ + t;
  }
  __syncthreads();
  {  // build e*tok tile (bf16) + norm partials; thread = (row, 64-elem segment)
    int row = tid >> 3;
    int d0 = (tid & 7) * 64;
    const float* er = emo + rJ[row] * D_;
    const float* cr = cau + rC[row] * D_;
    float s = 0.0f;
    for (int i = 0; i < 64; ++i) {
      float e = er[d0 + i], tk = cr[d0 + i];
      As[row][d0 + i] = (bf16_t)(e * tk);
      float df = e - tk;
      s += df * df;
    }
    segs[row][tid & 7] = s;
  }
  __syncthreads();
  if (tid < 16) {
    float s = 0.0f;
    for (int i = 0; i < 8; ++i) s += segs[tid][i];
    nrm[tid] = sqrtf(s);
  }
  __syncthreads();

  int wave = tid >> 5, lane = tid & 31;
  int laneHi = lane >> 4;
  int mRow = lane & 15;
  for (int nt = wave * 8; nt < wave * 8 + 8; ++nt) {  // each wave: 8 N-tiles (128 cols)
    v8f acc = {};
    int n = nt * 16 + (lane & 15);
    const bf16_t* brow = Dw + (size_t)n * D_;  // N x K row-major, L2-resident
    for (int kk = 0; kk < D_; kk += 32) {
      int kb = kk + laneHi * 8;
      v16bf af = cat8(*(const v8bf*)&As[mRow][kb], *(const v8bf*)&As[mRow][kb + 16]);
      int kg = kk + laneHi * 8;
      v16bf bfg = cat8(*(const v8bf*)&brow[kg], *(const v8bf*)&brow[kg + 16]);
      acc = __builtin_amdgcn_wmma_f32_16x16x32_bf16(
          false, af, false, bfg, (short)0, acc, false, false);
    }
    float w3cn = W3w[(size_t)n * 1537 + 1024];
    float bn = W3b[n];
    for (int i = 0; i < 8; ++i) {
      int m = i + laneHi * 8;
      float v = acc[i] + emoA[rJ[m] * D_ + n] + cauB[rC[m] * D_ + n] + nrm[m] * w3cn + bn;
      Hs[m][n] = fmaxf(v, 0.0f);
    }
  }
  __syncthreads();
  // fused classifier: wave handles 4 rows
  for (int mi = 0; mi < 4; ++mi) {
    int m = wave * 4 + mi;
    float s0 = 0.0f, s1 = 0.0f;
    for (int d = lane; d < D_; d += 32) {
      float hv = Hs[m][d];
      s0 += hv * clsw[d];
      s1 += hv * clsw[D_ + d];
    }
    for (int off = 16; off; off >>= 1) {
      s0 += __shfl_xor(s0, off, 32);
      s1 += __shfl_xor(s1, off, 32);
    }
    if (lane == 0) {
      int r = r0 + m;
      if (r < M8) {
        float l0 = s0 + clsb[0], l1 = s1 + clsb[1];
        float mx = fmaxf(l0, l1);
        float lse = mx + logf(expf(l0 - mx) + expf(l1 - mx));
        out2[r * 2 + 0] = l0 - lse;
        out2[r * 2 + 1] = l1 - lse;
      }
    }
  }
}

// ---------------- host orchestration ----------------
extern "C" void kernel_launch(void* const* d_in, const int* in_sizes, int n_in,
                              void* d_out, int out_size, void* d_ws, size_t ws_size,
                              hipStream_t stream) {
  (void)in_sizes; (void)n_in; (void)ws_size;
  const float* text_emo = (const float*)d_in[0];
  const float* text_cau = (const float*)d_in[1];
  const int*   label_ck = (const int*)d_in[2];
  const int*   label3   = (const int*)d_in[3];
  const float* rep_w    = (const float*)d_in[6];
  const float* rep_b    = (const float*)d_in[7];
  const float* W_w      = (const float*)d_in[8];
  const float* W_b      = (const float*)d_in[9];
  const float* Wo_w     = (const float*)d_in[10];
  const float* Wo_b     = (const float*)d_in[11];
  const float* gru_wih  = (const float*)d_in[12];
  const float* gru_whh  = (const float*)d_in[13];
  const float* gru_bih  = (const float*)d_in[14];
  const float* gru_bhh  = (const float*)d_in[15];
  const float* W3_w     = (const float*)d_in[16];
  const float* W3_b     = (const float*)d_in[17];
  const float* cls_w    = (const float*)d_in[18];
  const float* cls_b    = (const float*)d_in[19];

  char* ws = (char*)d_ws;
  auto carve = [&](size_t bytes) {
    char* p = ws;
    ws += (bytes + 255) & ~(size_t)255;
    return p;
  };
  bf16_t* rep_wb = (bf16_t*)carve(512 * 512 * 2);
  bf16_t* wih_b  = (bf16_t*)carve(1536 * 512 * 2);
  bf16_t* whh_b  = (bf16_t*)carve(1536 * 512 * 2);
  bf16_t* W1_b   = (bf16_t*)carve(512 * 512 * 2);
  bf16_t* W2_b   = (bf16_t*)carve(512 * 512 * 2);
  bf16_t* W3A_b  = (bf16_t*)carve(512 * 512 * 2);
  bf16_t* W3B_b  = (bf16_t*)carve(512 * 512 * 2);
  bf16_t* W3D_b  = (bf16_t*)carve(512 * 512 * 2);
  bf16_t* teb    = (bf16_t*)carve(256 * 512 * 2);
  bf16_t* tcb    = (bf16_t*)carve(512 * 512 * 2);
  bf16_t* emo_b  = (bf16_t*)carve(256 * 512 * 2);
  bf16_t* cau_b  = (bf16_t*)carve(512 * 512 * 2);
  bf16_t* h_b    = (bf16_t*)carve(64 * 512 * 2);
  float*  emo    = (float*)carve(256 * 512 * 4);
  float*  cau    = (float*)carve(512 * 512 * 4);
  float*  gx     = (float*)carve(64 * 1536 * 4);
  float*  gh     = (float*)carve(64 * 1536 * 4);
  float*  h      = (float*)carve(64 * 512 * 4);
  float*  emoP   = (float*)carve(256 * 512 * 4);
  float*  chP    = (float*)carve(64 * 512 * 4);
  float*  emoA   = (float*)carve(256 * 512 * 4);
  float*  cauB   = (float*)carve(512 * 512 * 4);
  int*    pj     = (int*)carve(16384 * 4);
  int*    pk     = (int*)carve(16384 * 4);
  int*    pcount = (int*)carve(4);

  auto cvt = [&](const float* src, int rowLen, int colOff, bf16_t* dst, int rows, int cols) {
    int n = rows * cols;
    cvt_bf16<<<(n + 255) / 256, 256, 0, stream>>>(src, rowLen, colOff, dst, rows, cols);
  };
  // weight / input conversions to bf16
  cvt(rep_w,    512,  0,    rep_wb, 512,  512);
  cvt(gru_wih,  512,  0,    wih_b,  1536, 512);
  cvt(gru_whh,  512,  0,    whh_b,  1536, 512);
  cvt(W_w,      1024, 0,    W1_b,   512,  512);
  cvt(W_w,      1024, 512,  W2_b,   512,  512);
  cvt(W3_w,     1537, 0,    W3A_b,  512,  512);
  cvt(W3_w,     1537, 512,  W3B_b,  512,  512);
  cvt(W3_w,     1537, 1025, W3D_b,  512,  512);
  cvt(text_emo, 512,  0,    teb,    256,  512);
  cvt(text_cau, 512,  0,    tcb,    512,  512);

  // emo/cau = relu(x @ rep_w^T + rep_b), f32 + bf16
  wmma_gemm<<<dim3(8, 4), 128, 0, stream>>>(teb, 512, rep_wb, rep_b, emo, emo_b, 256, 512, 512, 1);
  wmma_gemm<<<dim3(8, 8), 128, 0, stream>>>(tcb, 512, rep_wb, rep_b, cau, cau_b, 512, 512, 512, 1);

  // GRU over 8 time steps (x_t = cau rows k*8+t)
  zero_h<<<(64 * 512 + 255) / 256, 256, 0, stream>>>(h, h_b);
  for (int t = 0; t < CS_; ++t) {
    wmma_gemm<<<dim3(24, 1), 128, 0, stream>>>(cau_b + t * 512, CS_ * 512, wih_b, gru_bih,
                                               gx, nullptr, 64, 1536, 512, 0);
    wmma_gemm<<<dim3(24, 1), 128, 0, stream>>>(h_b, 512, whh_b, gru_bhh,
                                               gh, nullptr, 64, 1536, 512, 0);
    gru_gate<<<(64 * 512 + 255) / 256, 256, 0, stream>>>(gx, gh, h, h_b);
  }

  // phase2 factored: hidden = sigmoid(emoP[j] + chP[k]) (bias folded into emoP)
  wmma_gemm<<<dim3(8, 4), 128, 0, stream>>>(emo_b, 512, W1_b, W_b, emoP, nullptr, 256, 512, 512, 0);
  wmma_gemm<<<dim3(8, 1), 128, 0, stream>>>(h_b, 512, W2_b, nullptr, chP, nullptr, 64, 512, 512, 0);
  phase2_kernel<<<(SE_ * K_) / 8, 256, 0, stream>>>(emoP, chP, Wo_w, Wo_b, (float*)d_out);

  // pair selection + reusable precomputes for the pair GEMM
  pair_scan<<<1, 1024, 0, stream>>>(label_ck, pj, pk, pcount);
  wmma_gemm<<<dim3(8, 4), 128, 0, stream>>>(emo_b, 512, W3A_b, nullptr, emoA, nullptr, 256, 512, 512, 0);
  wmma_gemm<<<dim3(8, 8), 128, 0, stream>>>(cau_b, 512, W3B_b, nullptr, cauB, nullptr, 512, 512, 512, 0);

  // P derived from out_size: out = 32768 (phase2) + P*16 (logits) + P*8 (labels)
  int P = (out_size - 2 * SE_ * K_) / 24;
  if (P < 0) P = 0;
  int M8 = P * CS_;
  float* out2 = (float*)d_out + 2 * SE_ * K_;
  float* out3 = out2 + (size_t)P * 2 * CS_;
  if (M8 > 0) {
    pair_kernel<<<(M8 + 15) / 16, 128, 0, stream>>>(pj, pk, emo, cau, emoA, cauB,
                                                    W3D_b, W3_w, W3_b, cls_w, cls_b,
                                                    out2, M8);
    label3_kernel<<<(M8 + 255) / 256, 256, 0, stream>>>(pj, pk, label3, out3, M8);
  }
}
